// TransformerLayer_59459527246016
// MI455X (gfx1250) — compile-verified, matrix operand order checked by reading, not run on hardware
//
#include <hip/hip_runtime.h>

// ---------------- problem constants ----------------
constexpr int kB   = 2;
constexpr int kBS  = 1024;
constexpr int kWS  = 512;
constexpr int kS   = kBS + kWS;   // 1536
constexpr int kDIN = 2048;
constexpr int kH   = 16;
constexpr int kDH  = 128;
constexpr int kDM  = kH * kDH;    // 2048
constexpr int kDFF = 8192;

typedef unsigned short u16;
typedef __attribute__((ext_vector_type(16))) __bf16 v16bf;
typedef __attribute__((ext_vector_type(8)))  float  v8f;
typedef __attribute__((ext_vector_type(4)))  unsigned int u32x4;
typedef __attribute__((ext_vector_type(8)))  int i32x8;
typedef __attribute__((ext_vector_type(4)))  int i32x4;

#if defined(__has_builtin)
#if __has_builtin(__builtin_amdgcn_tensor_load_to_lds) && __has_builtin(__builtin_amdgcn_s_wait_tensorcnt)
#define USE_TDM 1
#endif
#endif
#ifndef USE_TDM
#define USE_TDM 0
#endif

__device__ __forceinline__ u16 f2bf(float f) {
    unsigned u = __float_as_uint(f);
    u += 0x7fffu + ((u >> 16) & 1u);       // round-to-nearest-even
    return (u16)(u >> 16);
}

__device__ __forceinline__ v16bf ldfrag(const u16* p) {
    v16bf r;
    __builtin_memcpy(&r, p, 32);
    return r;
}

__device__ __forceinline__ v8f wmma_bf16(v16bf a, v16bf b, v8f c) {
    return __builtin_amdgcn_wmma_f32_16x16x32_bf16(false, a, false, b, (short)0, c, false, false);
}

// ---------------- elementwise kernels ----------------

__global__ void f32_to_bf16_kernel(const float* __restrict__ in, u16* __restrict__ out, size_t n) {
    size_t i = (size_t)blockIdx.x * blockDim.x + threadIdx.x;
    size_t st = (size_t)gridDim.x * blockDim.x;
    for (; i < n; i += st) out[i] = f2bf(in[i]);
}

// rowwise LayerNorm(in)*g -> bf16 out  (rows of 2048, 256 threads/row)
__global__ __launch_bounds__(256) void ln_to_bf16_kernel(const float* __restrict__ in,
                                                         const float* __restrict__ g,
                                                         u16* __restrict__ out) {
    __shared__ float2 red[256];
    const int tid = threadIdx.x;
    const size_t row = blockIdx.x;
    const float* src = in + row * kDM;
    float v[8];
    float sm = 0.f, sq = 0.f;
#pragma unroll
    for (int e = 0; e < 8; ++e) {
        v[e] = src[tid * 8 + e];
        sm += v[e];
        sq += v[e] * v[e];
    }
    red[tid] = make_float2(sm, sq);
    __syncthreads();
    for (int s = 128; s > 0; s >>= 1) {
        if (tid < s) { red[tid].x += red[tid + s].x; red[tid].y += red[tid + s].y; }
        __syncthreads();
    }
    const float mean = red[0].x * (1.f / kDM);
    const float var  = red[0].y * (1.f / kDM) - mean * mean;
    const float istd = rsqrtf(var + 1e-5f);
#pragma unroll
    for (int e = 0; e < 8; ++e) {
        const int c = tid * 8 + e;
        out[row * kDM + c] = f2bf((v[e] - mean) * istd * g[c]);
    }
}

// optional-LN + RoPE -> bf16.  in: (B*rpb, DM) fp32.
__global__ __launch_bounds__(256) void ln_rope_kernel(const float* __restrict__ in,
                                                      const float* __restrict__ g,
                                                      const float* __restrict__ rot,
                                                      const float* __restrict__ xsc,
                                                      u16* __restrict__ out,
                                                      int rpb, int rot_off, int ln_from, int inv) {
    __shared__ float2 red[256];
    const int tid = threadIdx.x;
    const int row = blockIdx.x;
    const int s = (row >= rpb) ? (row - rpb) : row;    // batch count is 2
    const bool doLN = (s >= ln_from);                  // block-uniform
    const float* src = in + (size_t)row * kDM;
    float v[8];
#pragma unroll
    for (int e = 0; e < 8; ++e) v[e] = src[tid * 8 + e];
    float mean = 0.f, istd = 1.f;
    if (doLN) {
        float sm = 0.f, sq = 0.f;
#pragma unroll
        for (int e = 0; e < 8; ++e) { sm += v[e]; sq += v[e] * v[e]; }
        red[tid] = make_float2(sm, sq);
        __syncthreads();
        for (int st = 128; st > 0; st >>= 1) {
            if (tid < st) { red[tid].x += red[tid + st].x; red[tid].y += red[tid + st].y; }
            __syncthreads();
        }
        mean = red[0].x * (1.f / kDM);
        const float var = red[0].y * (1.f / kDM) - mean * mean;
        istd = rsqrtf(var + 1e-5f);
    }
    const size_t rr = (size_t)(rot_off + s) * kDM;
#pragma unroll
    for (int pr = 0; pr < 4; ++pr) {
        const int c0 = tid * 8 + pr * 2;
        float xe = v[pr * 2], xo = v[pr * 2 + 1];
        if (doLN) {
            xe = (xe - mean) * istd * g[c0];
            xo = (xo - mean) * istd * g[c0 + 1];
        }
        const float fe = rot[rr + c0], fo = rot[rr + c0 + 1];
        float se = xsc[rr + c0], so = xsc[rr + c0 + 1];
        if (inv) { se = 1.f / se; so = 1.f / so; }
        const float oe = xe * cosf(fe) * se - xo * sinf(fe) * se;
        const float oo = xo * cosf(fo) * so + xe * sinf(fo) * so;
        out[(size_t)row * kDM + c0]     = f2bf(oe);
        out[(size_t)row * kDM + c0 + 1] = f2bf(oo);
    }
}

// splice caches: cache_k -> k_full fp32 rows [b*S, b*S+WS), cache_v -> v_b bf16 same rows
__global__ void cache_copy_kernel(const float* __restrict__ ck, const float* __restrict__ cv,
                                  float* __restrict__ kf, u16* __restrict__ vb) {
    const int row = blockIdx.x;              // 0 .. B*WS-1
    const int b = row >> 9, s = row & (kWS - 1);
    const size_t srow = (size_t)row * kDM;
    const size_t drow = ((size_t)b * kS + s) * kDM;
    for (int c = threadIdx.x; c < kDM; c += blockDim.x) {
        kf[drow + c] = ck[srow + c];
        vb[drow + c] = f2bf(cv[srow + c]);
    }
}

// ---------------- tiled NT GEMM with WMMA bf16 + TDM double buffering ----------------
// C(M,N) = A(M,K) * W(N,K)^T   (both bf16, K contiguous) with flexible epilogue.
// mode 0: fp32, row -> (roff + b*bstr + s), plain cols
// mode 1: bf16, same row map (optional relu)
// mode 2: bf16 head layout   out[((b*H + n/DH)*rpb + s)*DH + n%DH]
// mode 3: bf16 head^T layout out[((b*H + n/DH)*DH + n%DH)*rpb + s]
#define GMT 128
#define GNT 128
#define GKT 32
#define LDP 40   // padded LDS row stride (elements): 64B data + 16B pad

#if USE_TDM
// Issue one TDM descriptor: 2D tile (GKT x 128 rows), 2-byte elements,
// LDS padding of 4 DWORDs (16B) after every 16 DWORDs (64B) -> 40-element rows.
__device__ __forceinline__ void tdm_load_tile(const u16* gsrc, unsigned lds_addr, int rows, int ldk) {
    const unsigned long long ga = (unsigned long long)gsrc;
    u32x4 g0;
    g0[0] = 1u;                                              // count=1, user mode
    g0[1] = lds_addr;                                        // lds byte address
    g0[2] = (unsigned)ga;                                    // global addr [31:0]
    g0[3] = ((unsigned)(ga >> 32) & 0x01FFFFFFu) | (2u << 30); // ga[56:32] | type=2
    i32x8 g1;
    g1[0] = (int)((1u << 16) | (1u << 20) | (3u << 22) | (3u << 25)); // data_size=2B, pad_en, int=16dw, amt=4dw
    g1[1] = (int)((unsigned)ldk << 16);                      // tensor_dim0[15:0] in [31:16]
    g1[2] = (int)((unsigned)rows << 16);                     // tensor_dim1[15:0] in [31:16]
    g1[3] = (int)((unsigned)GKT << 16);                      // tile_dim0=32 in [31:16]
    g1[4] = GMT;                                             // tile_dim1=128 ; tile_dim2=0
    g1[5] = ldk;                                             // tensor_dim0_stride[31:0]
    g1[6] = 0;
    g1[7] = 0;
    i32x4 z4 = {0, 0, 0, 0};
    i32x8 z8 = {0, 0, 0, 0, 0, 0, 0, 0};
    __builtin_amdgcn_tensor_load_to_lds(g0, g1, z4, z4, z8, 0);
}
#endif

__global__ __launch_bounds__(256) void gemm_bf16_wmma(const u16* __restrict__ A,
                                                      const u16* __restrict__ W,
                                                      const float* __restrict__ bias,
                                                      float* __restrict__ outF,
                                                      u16* __restrict__ outB,
                                                      int M, int N, int K,
                                                      int mode, int relu,
                                                      int rpb, int bstr, int roff) {
    __shared__ u16 ldsA[2][GMT * LDP];
    __shared__ u16 ldsB[2][GNT * LDP];
    const int tid  = threadIdx.x;
    const int lane = tid & 31, w = tid >> 5;
    const int wm = w & 3, wn = w >> 2;         // 4 (M) x 2 (N) waves; wave tile 32x64
    const int lrow = lane & 15, lhalf = lane >> 4;
    const int m0 = blockIdx.y * GMT, n0 = blockIdx.x * GNT;

    v8f acc[2][4];
#pragma unroll
    for (int ai = 0; ai < 2; ++ai)
#pragma unroll
        for (int bj = 0; bj < 4; ++bj)
#pragma unroll
            for (int e = 0; e < 8; ++e) acc[ai][bj][e] = 0.f;

    const int iters = K / GKT;

#if USE_TDM
    const unsigned aAddr[2] = { (unsigned)(unsigned long long)&ldsA[0][0],
                                (unsigned)(unsigned long long)&ldsA[1][0] };
    const unsigned bAddr[2] = { (unsigned)(unsigned long long)&ldsB[0][0],
                                (unsigned)(unsigned long long)&ldsB[1][0] };
    if (w == 0) {
        tdm_load_tile(A + (size_t)m0 * K, aAddr[0], M, K);
        tdm_load_tile(W + (size_t)n0 * K, bAddr[0], N, K);
    }
#else
    const int ldr = tid >> 1;                  // 0..127 tile row
    const int ldh = tid & 1;                   // K half
#endif

    for (int it = 0; it < iters; ++it) {
        const int buf = it & 1;
#if USE_TDM
        if (w == 0) {
            if (it + 1 < iters) {
                const int k1 = (it + 1) * GKT;
                tdm_load_tile(A + (size_t)m0 * K + k1, aAddr[buf ^ 1], M, K);
                tdm_load_tile(W + (size_t)n0 * K + k1, bAddr[buf ^ 1], N, K);
                __builtin_amdgcn_s_wait_tensorcnt(2);   // current tile complete
            } else {
                __builtin_amdgcn_s_wait_tensorcnt(0);
            }
        }
        __syncthreads();
#else
        {
            const int k0 = it * GKT;
            const uint4* pa = (const uint4*)(A + (size_t)(m0 + ldr) * K + k0 + ldh * 16);
            uint4 a0 = pa[0], a1 = pa[1];
            const uint4* pw = (const uint4*)(W + (size_t)(n0 + ldr) * K + k0 + ldh * 16);
            uint4 w0 = pw[0], w1 = pw[1];
            uint4* da = (uint4*)&ldsA[buf][ldr * LDP + ldh * 16];
            da[0] = a0; da[1] = a1;
            uint4* db = (uint4*)&ldsB[buf][ldr * LDP + ldh * 16];
            db[0] = w0; db[1] = w1;
        }
        __syncthreads();
#endif
        v16bf af[2], bfr[4];
#pragma unroll
        for (int ai = 0; ai < 2; ++ai)
            af[ai] = ldfrag(&ldsA[buf][(wm * 32 + ai * 16 + lrow) * LDP + lhalf * 16]);
#pragma unroll
        for (int bj = 0; bj < 4; ++bj)
            bfr[bj] = ldfrag(&ldsB[buf][(wn * 64 + bj * 16 + lrow) * LDP + lhalf * 16]);
#pragma unroll
        for (int ai = 0; ai < 2; ++ai)
#pragma unroll
            for (int bj = 0; bj < 4; ++bj)
                acc[ai][bj] = wmma_bf16(af[ai], bfr[bj], acc[ai][bj]);
        __syncthreads();
    }

#pragma unroll
    for (int ai = 0; ai < 2; ++ai) {
#pragma unroll
        for (int bj = 0; bj < 4; ++bj) {
            const int col = n0 + wn * 64 + bj * 16 + lrow;
            const float bv = bias ? bias[col] : 0.f;
#pragma unroll
            for (int r = 0; r < 8; ++r) {
                const int row = m0 + wm * 32 + ai * 16 + lhalf * 8 + r;
                float v = acc[ai][bj][r] + bv;
                if (relu) v = fmaxf(v, 0.f);
                const int bb = (row >= rpb) ? 1 : 0;     // batch count <= 2
                const int s = row - (bb ? rpb : 0);
                if (mode == 0) {
                    outF[((size_t)(roff + bb * bstr + s)) * N + col] = v;
                } else if (mode == 1) {
                    outB[((size_t)(roff + bb * bstr + s)) * N + col] = f2bf(v);
                } else if (mode == 2) {
                    outB[(((size_t)(bb * kH + (col >> 7))) * rpb + s) * kDH + (col & 127)] = f2bf(v);
                } else {
                    outB[(((size_t)(bb * kH + (col >> 7))) * kDH + (col & 127)) * (size_t)rpb + s] = f2bf(v);
                }
            }
        }
    }
}

// ---------------- flash attention (one wave per 16-row Q tile) ----------------
// qh: (B,H,BS,DH) bf16 ; kh: (B,H,S,DH) bf16 ; vhT: (B,H,DH,S) bf16
// out attn: (B*BS, DM) bf16, head h at columns [h*DH, (h+1)*DH)
__global__ __launch_bounds__(32) void flash_kernel(const u16* __restrict__ qh,
                                                   const u16* __restrict__ kh,
                                                   const u16* __restrict__ vhT,
                                                   u16* __restrict__ attn) {
    const int lane = threadIdx.x;
    const int bid = blockIdx.x;
    const int qt = bid & 63;              // BS/16 tiles
    const int h  = (bid >> 6) & (kH - 1);
    const int b  = bid >> 10;
    const int i0 = qt * 16;
    const int lrow = lane & 15, lhalf = lane >> 4;

    const u16* qbase = qh  + ((size_t)(b * kH + h) * kBS) * kDH;
    const u16* kbase = kh  + ((size_t)(b * kH + h) * kS)  * kDH;
    const u16* vbase = vhT + ((size_t)(b * kH + h) * kDH) * kS;

    v16bf aQ[4];
#pragma unroll
    for (int kd = 0; kd < 4; ++kd)
        aQ[kd] = ldfrag(qbase + (size_t)(i0 + lrow) * kDH + kd * 32 + lhalf * 16);

    v8f O[8];
    float m[8], l[8];
#pragma unroll
    for (int dj = 0; dj < 8; ++dj)
#pragma unroll
        for (int e = 0; e < 8; ++e) O[dj][e] = 0.f;
#pragma unroll
    for (int r = 0; r < 8; ++r) { m[r] = -3e30f; l[r] = 0.f; }

    __shared__ u16 Pt[16 * 72];
    const float sc = 0.0883883476483184f;     // 1/sqrt(128)

    for (int kb = 0; kb < kS / 64; ++kb) {
        const int n0 = kb * 64;
        v8f Sf[4];
#pragma unroll
        for (int kj = 0; kj < 4; ++kj)
#pragma unroll
            for (int e = 0; e < 8; ++e) Sf[kj][e] = 0.f;

#pragma unroll
        for (int kd = 0; kd < 4; ++kd) {
#pragma unroll
            for (int kj = 0; kj < 4; ++kj) {
                v16bf bk = ldfrag(kbase + (size_t)(n0 + kj * 16 + lrow) * kDH + kd * 32 + lhalf * 16);
                Sf[kj] = wmma_bf16(aQ[kd], bk, Sf[kj]);
            }
        }

        float nm[8];
#pragma unroll
        for (int r = 0; r < 8; ++r) nm[r] = m[r];
#pragma unroll
        for (int kj = 0; kj < 4; ++kj) {
            const int j = n0 + kj * 16 + lrow;
#pragma unroll
            for (int r = 0; r < 8; ++r) {
                const int i = i0 + lhalf * 8 + r;
                float v = Sf[kj][r] * sc;
                if (!((j > i) && (j <= i + kWS))) v = -1e9f;
                Sf[kj][r] = v;
                nm[r] = fmaxf(nm[r], v);
            }
        }
#pragma unroll
        for (int off = 1; off < 16; off <<= 1)
#pragma unroll
            for (int r = 0; r < 8; ++r)
                nm[r] = fmaxf(nm[r], __shfl_xor(nm[r], off, 32));

        float alpha[8];
#pragma unroll
        for (int r = 0; r < 8; ++r) {
            alpha[r] = __expf(m[r] - nm[r]);
            m[r] = nm[r];
            l[r] *= alpha[r];
        }
#pragma unroll
        for (int dj = 0; dj < 8; ++dj)
#pragma unroll
            for (int r = 0; r < 8; ++r) O[dj][r] *= alpha[r];

        float rs[8];
#pragma unroll
        for (int r = 0; r < 8; ++r) rs[r] = 0.f;
#pragma unroll
        for (int kj = 0; kj < 4; ++kj)
#pragma unroll
            for (int r = 0; r < 8; ++r) {
                const float p = __expf(Sf[kj][r] - m[r]);
                Sf[kj][r] = p;
                rs[r] += p;
            }
#pragma unroll
        for (int off = 1; off < 16; off <<= 1)
#pragma unroll
            for (int r = 0; r < 8; ++r) rs[r] += __shfl_xor(rs[r], off, 32);
#pragma unroll
        for (int r = 0; r < 8; ++r) l[r] += rs[r];

        // stage P (bf16) through LDS: transpose C-layout -> A-fragment layout
#pragma unroll
        for (int kj = 0; kj < 4; ++kj)
#pragma unroll
            for (int r = 0; r < 8; ++r)
                Pt[(lhalf * 8 + r) * 72 + kj * 16 + lrow] = f2bf(Sf[kj][r]);
        __syncthreads();

#pragma unroll
        for (int ks = 0; ks < 2; ++ks) {
            v16bf aP = ldfrag(&Pt[lrow * 72 + ks * 32 + lhalf * 16]);
#pragma unroll
            for (int dj = 0; dj < 8; ++dj) {
                v16bf bv = ldfrag(vbase + (size_t)(dj * 16 + lrow) * kS + n0 + ks * 32 + lhalf * 16);
                O[dj] = wmma_bf16(aP, bv, O[dj]);
            }
        }
        __syncthreads();
    }

#pragma unroll
    for (int dj = 0; dj < 8; ++dj) {
#pragma unroll
        for (int r = 0; r < 8; ++r) {
            const int i = i0 + lhalf * 8 + r;
            const int c = h * kDH + dj * 16 + lrow;
            attn[((size_t)(b * kBS + i)) * kDM + c] = f2bf(O[dj][r] / l[r]);
        }
    }
}

// ---------------- host orchestration ----------------
extern "C" void kernel_launch(void* const* d_in, const int* in_sizes, int n_in,
                              void* d_out, int out_size, void* d_ws, size_t ws_size,
                              hipStream_t stream) {
    const float* x     = (const float*)d_in[0];
    const float* ck    = (const float*)d_in[1];
    const float* cv    = (const float*)d_in[2];
    const float* rot   = (const float*)d_in[3];
    const float* xsc   = (const float*)d_in[4];
    const float* g_in  = (const float*)d_in[6];
    const float* Wq    = (const float*)d_in[7];
    const float* Wk    = (const float*)d_in[8];
    const float* Wv    = (const float*)d_in[9];
    const float* g_q   = (const float*)d_in[10];
    const float* g_k   = (const float*)d_in[11];
    const float* Wlq   = (const float*)d_in[12];
    const float* blq   = (const float*)d_in[13];
    const float* Wlk   = (const float*)d_in[14];
    const float* blk   = (const float*)d_in[15];
    const float* Wlv   = (const float*)d_in[16];
    const float* blv   = (const float*)d_in[17];
    const float* Wo    = (const float*)d_in[18];
    const float* bo    = (const float*)d_in[19];
    const float* g_ffn = (const float*)d_in[20];
    const float* W1    = (const float*)d_in[21];
    const float* W2    = (const float*)d_in[22];
    float* out = (float*)d_out;

    char* p = (char*)d_ws;
    auto alloc = [&](size_t bytes) -> void* {
        void* r = (void*)p;
        p += (bytes + 255) & ~(size_t)255;
        return r;
    };
    const size_t E44 = (size_t)kDM * kDIN;        // 4M (2048x2048)
    const size_t Eff = (size_t)kDFF * kDM;        // 16M
    u16* Wq_b  = (u16*)alloc(E44 * 2);
    u16* Wk_b  = (u16*)alloc(E44 * 2);
    u16* Wv_b  = (u16*)alloc(E44 * 2);
    u16* Wlq_b = (u16*)alloc(E44 * 2);
    u16* Wlk_b = (u16*)alloc(E44 * 2);
    u16* Wlv_b = (u16*)alloc(E44 * 2);
    u16* Wo_b  = (u16*)alloc(E44 * 2);
    u16* W1_b  = (u16*)alloc(Eff * 2);
    u16* W2_b  = (u16*)alloc(Eff * 2);

    const size_t rowsQ = (size_t)kB * kBS;        // 2048
    const size_t rowsK = (size_t)kB * kS;         // 3072
    u16*   xn_b   = (u16*)  alloc(rowsQ * kDM * 2);
    float* q_pre  = (float*)alloc(rowsQ * kDM * 4);
    float* k_full = (float*)alloc(rowsK * kDM * 4);
    u16*   v_b    = (u16*)  alloc(rowsK * kDM * 2);
    u16*   q_r    = (u16*)  alloc(rowsQ * kDM * 2);
    u16*   k_r    = (u16*)  alloc(rowsK * kDM * 2);
    u16*   qh     = (u16*)  alloc(rowsQ * kDM * 2);
    u16*   kh     = (u16*)  alloc(rowsK * kDM * 2);
    u16*   vhT    = (u16*)  alloc(rowsK * kDM * 2);
    u16*   attn   = (u16*)  alloc(rowsQ * kDM * 2);
    float* o_f    = (float*)alloc(rowsQ * kDM * 4);
    u16*   hln    = (u16*)  alloc(rowsQ * kDM * 2);
    u16*   h1     = (u16*)  alloc((size_t)rowsQ * kDFF * 2);

    auto conv = [&](const float* src, u16* dst, size_t n) {
        f32_to_bf16_kernel<<<dim3(2048), dim3(256), 0, stream>>>(src, dst, n);
    };
    auto gemm = [&](const u16* A, const u16* Wt, const float* bias, float* oF, u16* oB,
                    int M, int N, int K, int mode, int relu, int rpb, int bstr, int roff) {
        dim3 g(N / GNT, M / GMT);
        gemm_bf16_wmma<<<g, dim3(256), 0, stream>>>(A, Wt, bias, oF, oB, M, N, K,
                                                    mode, relu, rpb, bstr, roff);
    };

    // 0) weights -> bf16 (all fit in L2 afterwards: ~120 MB < 192 MB)
    conv(Wq, Wq_b, E44);  conv(Wk, Wk_b, E44);  conv(Wv, Wv_b, E44);
    conv(Wlq, Wlq_b, E44); conv(Wlk, Wlk_b, E44); conv(Wlv, Wlv_b, E44);
    conv(Wo, Wo_b, E44);  conv(W1, W1_b, Eff);  conv(W2, W2_b, Eff);

    // 1) xn = LN(x)*g_in  -> bf16
    ln_to_bf16_kernel<<<dim3((unsigned)rowsQ), dim3(256), 0, stream>>>(x, g_in, xn_b);

    // 2) QKV projections
    gemm(xn_b, Wq_b, nullptr, q_pre, nullptr, kB * kBS, kDM, kDIN, 0, 0, kB * kBS, 0, 0);
    gemm(xn_b, Wk_b, nullptr, k_full, nullptr, kB * kBS, kDM, kDIN, 0, 0, kBS, kS, kWS);
    gemm(xn_b, Wv_b, nullptr, nullptr, v_b, kB * kBS, kDM, kDIN, 1, 0, kBS, kS, kWS);

    // 3) splice caches into k_full (fp32) and v_b (bf16)
    cache_copy_kernel<<<dim3(kB * kWS), dim3(256), 0, stream>>>(ck, cv, k_full, v_b);

    // 4) LN + RoPE
    ln_rope_kernel<<<dim3((unsigned)rowsQ), dim3(256), 0, stream>>>(q_pre, g_q, rot, xsc, q_r,
                                                                    kBS, kWS, 0, 0);
    ln_rope_kernel<<<dim3((unsigned)rowsK), dim3(256), 0, stream>>>(k_full, g_k, rot, xsc, k_r,
                                                                    kS, 0, kWS, 1);

    // 5) head projections (+bias), emitting attention-friendly layouts
    gemm(q_r, Wlq_b, blq, nullptr, qh,  kB * kBS, kDM, kDM, 2, 0, kBS, 0, 0);
    gemm(k_r, Wlk_b, blk, nullptr, kh,  kB * kS,  kDM, kDM, 2, 0, kS,  0, 0);
    gemm(v_b, Wlv_b, blv, nullptr, vhT, kB * kS,  kDM, kDM, 3, 0, kS,  0, 0);

    // 6) masked flash attention
    flash_kernel<<<dim3(kB * kH * (kBS / 16)), dim3(32), 0, stream>>>(qh, kh, vhT, attn);

    // 7) output projection + LN + FFN
    gemm(attn, Wo_b, bo, o_f, nullptr, kB * kBS, kDM, kDM, 0, 0, kB * kBS, 0, 0);
    ln_to_bf16_kernel<<<dim3((unsigned)rowsQ), dim3(256), 0, stream>>>(o_f, g_ffn, hln);
    gemm(hln, W1_b, nullptr, nullptr, h1, kB * kBS, kDFF, kDM, 1, 1, kB * kBS, 0, 0);
    gemm(h1, W2_b, nullptr, out, nullptr, kB * kBS, kDIN, kDFF, 0, 0, kB * kBS, 0, 0);
}
